// MyGRULanguageModel_40827959115851
// MI455X (gfx1250) — matrix operand order repaired and checked by reading, not compiled, not run on hardware
//
#include <hip/hip_runtime.h>

#define VOCAB   50257
#define D_MODEL 256
#define HIDDEN  512
#define GATES   1536   // 3*HIDDEN
#define BATCH   64
#define SEQ     512

typedef __bf16 bf16;
typedef __attribute__((ext_vector_type(16))) bf16  v16bf;
typedef __attribute__((ext_vector_type(8)))  bf16  bf16x8;
typedef __attribute__((ext_vector_type(8)))  float v8f;
typedef unsigned int u32x4 __attribute__((ext_vector_type(4)));
typedef int          i32x8 __attribute__((ext_vector_type(8)));
typedef int          i32x4 __attribute__((ext_vector_type(4)));

#if defined(__has_builtin)
#if __has_builtin(__builtin_amdgcn_tensor_load_to_lds) && __has_builtin(__builtin_amdgcn_s_wait_tensorcnt)
#define HAVE_TDM 1
#endif
#endif

// ---------------- workspace layout (bytes) ----------------
#define OFF_WXB   ((size_t)0)                         // 256*1536*2   = 786432
#define OFF_WHB   ((size_t)786432)                    // 512*1536*2   = 1572864
#define OFF_XG    ((size_t)2359296)                   // 512*64*1536*2 = 100663296
#define OFF_HBUF  ((size_t)103022592)                 // 2*64*512*2   = 131072
#define OFF_HLAST ((size_t)103153664)                 // 64*512*4     = 131072
#define OFF_SYNC  ((size_t)103284736)                 // 1024
#define WS_NEEDED ((size_t)103285760)

// ---------------- device helpers ----------------
__device__ __forceinline__ v16bf mk16(bf16x8 lo, bf16x8 hi) {
  v16bf r;
#pragma unroll
  for (int i = 0; i < 8; ++i) { r[i] = lo[i]; r[i + 8] = hi[i]; }
  return r;
}

__device__ __forceinline__ v8f wmma_bf16(v16bf a, v16bf b, v8f c) {
  // D(f32,16x16) = A(bf16,16x32) * B(bf16,32x16) + C   -> v_wmma_f32_16x16x32_bf16
  return __builtin_amdgcn_wmma_f32_16x16x32_bf16(false, a, false, b, (short)0, c, false, false);
}

__device__ __forceinline__ float sigmoidf_(float x) {
  return 1.0f / (1.0f + __expf(-x));
}

#ifdef HAVE_TDM
// 2D bf16 tile load Global->LDS via Tensor Data Mover (D# per CDNA5 ISA ch.8).
// 6-arg builtin form: (u32x4 g0, i32x8 g1, i32x4 g2, i32x4 g3, i32x8 g4, i32 cpol)
__device__ __forceinline__ void tdm_load_2d_bf16(unsigned lds_byte, const void* gptr,
                                                 unsigned width, unsigned rows,
                                                 unsigned row_stride) {
  unsigned long long ga = (unsigned long long)(uintptr_t)gptr;
  u32x4 g0;
  g0[0] = 1u;                                        // count=1, user descriptor
  g0[1] = lds_byte;                                  // lds_addr
  g0[2] = (unsigned)(ga & 0xffffffffull);            // global_addr[31:0]
  g0[3] = (unsigned)((ga >> 32) & 0x1ffffffull) | (2u << 30);  // addr[56:32], type=2
  i32x8 g1;
  g1[0] = 0x00010000;                                // data_size=1 (2 bytes), mask=0
  g1[1] = (int)((width & 0xffffu) << 16);            // tensor_dim0[15:0]
  g1[2] = (int)(((width >> 16) & 0xffffu) | ((rows & 0xffffu) << 16));   // dim0 hi | dim1 lo
  g1[3] = (int)(((rows >> 16) & 0xffffu) | ((width & 0xffffu) << 16));   // dim1 hi | tile_dim0
  g1[4] = (int)(rows & 0xffffu);                     // tile_dim1, tile_dim2=0
  g1[5] = (int)row_stride;                           // tensor_dim0_stride lo
  g1[6] = 0;                                         // stride0 hi | stride1 lo
  g1[7] = 0;
  i32x4 z4; z4[0] = z4[1] = z4[2] = z4[3] = 0;
  i32x8 z8;
#pragma unroll
  for (int i = 0; i < 8; ++i) z8[i] = 0;
  __builtin_amdgcn_tensor_load_to_lds(g0, g1, z4, z4, z8, 0);
}
#endif

__device__ __forceinline__ void group_barrier(int* cnt, int* gen, int members) {
  __syncthreads();
  if (threadIdx.x == 0) {
    __threadfence();
    volatile int* vgen = (volatile int*)gen;
    int old = *vgen;
    if (atomicAdd(cnt, 1) == members - 1) {
      atomicExch(cnt, 0);
      __threadfence();
      atomicAdd(gen, 1);
    } else {
      while (*vgen == old) __builtin_amdgcn_s_sleep(2);
    }
    __threadfence();
  }
  __syncthreads();
}

// ---------------- kernel 0: f32 -> bf16 weight conversion ----------------
__global__ void cvt_weights_kernel(const float* __restrict__ Wx, const float* __restrict__ Wh,
                                   bf16* __restrict__ Wxb, bf16* __restrict__ Whb) {
  const int n1 = D_MODEL * GATES;   // 393216
  const int n2 = HIDDEN * GATES;    // 786432
  for (int i = blockIdx.x * blockDim.x + threadIdx.x; i < n1 + n2;
       i += gridDim.x * blockDim.x) {
    if (i < n1) Wxb[i] = (bf16)Wx[i];
    else        Whb[i - n1] = (bf16)Wh[i - n1];
  }
}

// ---------------- kernel 1: xg = emb[ids] @ Wx + bx  (bf16 WMMA) ----------------
// grid.x = (B*S)/16 = 2048 M-tiles (m = t*64 + b), grid.y = 6 chunks of 256 cols.
// 128 threads = 4 waves; each wave: 4 N-tiles of 16.
__global__ __launch_bounds__(128, 1)
void xg_gemm_kernel(const int* __restrict__ ids, const float* __restrict__ emb,
                    const bf16* __restrict__ Wxb, const float* __restrict__ bx,
                    bf16* __restrict__ xg) {
  const int tid  = threadIdx.x;
  const int wave = tid >> 5;
  const int lane = tid & 31;
  const int hi   = lane >> 4;
  const int m    = lane & 15;          // A row / C col within tile
  const int mt   = blockIdx.x;
  const int t    = mt >> 2;
  const int bb0  = (mt & 3) * 16;
  const int nbase = blockIdx.y * 256 + wave * 64;

  const int b  = bb0 + m;
  const int id = ids[(size_t)b * SEQ + t];
  const float* arow = emb + (size_t)id * D_MODEL;

  v8f acc[4];
#pragma unroll
  for (int nt = 0; nt < 4; ++nt)
#pragma unroll
    for (int v = 0; v < 8; ++v) acc[nt][v] = 0.0f;

#pragma unroll
  for (int kk = 0; kk < D_MODEL / 32; ++kk) {
    // A fragment: row m of emb, K = kk*32 + hi*8 + {0..7, 16..23}, f32 -> bf16
    const float* ap = arow + kk * 32 + hi * 8;
    float4 f0 = *(const float4*)(ap + 0);
    float4 f1 = *(const float4*)(ap + 4);
    float4 f2 = *(const float4*)(ap + 16);
    float4 f3 = *(const float4*)(ap + 20);
    v16bf A;
    A[0]  = (bf16)f0.x; A[1]  = (bf16)f0.y; A[2]  = (bf16)f0.z; A[3]  = (bf16)f0.w;
    A[4]  = (bf16)f1.x; A[5]  = (bf16)f1.y; A[6]  = (bf16)f1.z; A[7]  = (bf16)f1.w;
    A[8]  = (bf16)f2.x; A[9]  = (bf16)f2.y; A[10] = (bf16)f2.z; A[11] = (bf16)f2.w;
    A[12] = (bf16)f3.x; A[13] = (bf16)f3.y; A[14] = (bf16)f3.z; A[15] = (bf16)f3.w;

    const int row = kk * 32 + hi * 8;   // B col m, K = row + {0..7, 16..23}
#pragma unroll
    for (int nt = 0; nt < 4; ++nt) {
      const bf16* bp = Wxb + (size_t)row * GATES + nbase + nt * 16 + m;
      v16bf Bv;
#pragma unroll
      for (int i = 0; i < 8; ++i) { Bv[i] = bp[(size_t)i * GATES]; Bv[i + 8] = bp[(size_t)(i + 16) * GATES]; }
      acc[nt] = wmma_bf16(A, Bv, acc[nt]);
    }
  }

#pragma unroll
  for (int nt = 0; nt < 4; ++nt) {
    const int col = nbase + nt * 16 + m;
    const float bxv = bx[col];
#pragma unroll
    for (int v = 0; v < 8; ++v) {
      const int row = mt * 16 + v + 8 * hi;   // == t*64 + batch
      xg[(size_t)row * GATES + col] = (bf16)(acc[nt][v] + bxv);
    }
  }
}

// ---------------- kernel 2: GRU scan ----------------
// 32 WGs x 128 thr. group = blockIdx/8 owns batches [group*16,+16);
// wg = blockIdx%8 owns hidden units [wg*64,+64) (3 gate column-blocks of Wh).
// Wh slice (3*512*64 bf16 = 192KB) lives in LDS for all 512 steps (TDM-staged).
__global__ __launch_bounds__(128, 1)
void gru_scan_kernel(const bf16* __restrict__ Whb, const bf16* __restrict__ xg,
                     const float* __restrict__ bh, bf16* __restrict__ hbuf,
                     float* __restrict__ hlast, int* __restrict__ sync) {
  extern __shared__ char smem[];
  bf16* wh_s = (bf16*)smem;                                  // [3][512][64]
  bf16* h_s  = (bf16*)(smem + 3 * HIDDEN * 64 * 2);          // [16][512]

  const int tid  = threadIdx.x;
  const int wave = tid >> 5;
  const int lane = tid & 31;
  const int hi   = lane >> 4;
  const int m    = lane & 15;
  const int group = blockIdx.x >> 3;
  const int wg    = blockIdx.x & 7;
  const int b0 = group * 16;
  const int j0 = wg * 64;
  int* cnt = sync + group * 64;
  int* gen = cnt + 16;

  // ---- stage Wh gate slices into LDS (once) ----
#ifdef HAVE_TDM
  if (tid < 32) {
    const unsigned lds_base = (unsigned)__builtin_amdgcn_groupstaticsize();
#pragma unroll
    for (int g = 0; g < 3; ++g)
      tdm_load_2d_bf16(lds_base + (unsigned)g * HIDDEN * 64u * 2u,
                       Whb + (size_t)g * HIDDEN + j0,
                       64u, (unsigned)HIDDEN, (unsigned)GATES);
  }
  __builtin_amdgcn_s_wait_tensorcnt(0);
#else
  for (int c = tid; c < 3 * HIDDEN * 8; c += 128) {
    const int g  = c / (HIDDEN * 8);
    const int r  = (c >> 3) % HIDDEN;
    const int c8 = (c & 7) * 8;
    *(bf16x8*)(wh_s + ((size_t)(g * HIDDEN + r) * 64 + c8)) =
        *(const bf16x8*)(Whb + (size_t)r * GATES + g * HIDDEN + j0 + c8);
  }
#endif
  __syncthreads();

  // per-lane constants: this lane's hidden column
  const int jl   = wave * 16 + m;       // 0..63 within WG slice
  const int jcol = j0 + jl;             // 0..511 global hidden index
  const float bhr = bh[jcol];
  const float bhz = bh[HIDDEN + jcol];
  const float bhn = bh[2 * HIDDEN + jcol];

  float hold[8];
#pragma unroll
  for (int v = 0; v < 8; ++v) hold[v] = 0.0f;

  for (int t = 0; t < SEQ; ++t) {
    const bf16* cur = hbuf + (size_t)(t & 1) * BATCH * HIDDEN;
    bf16* nxt       = hbuf + (size_t)((t + 1) & 1) * BATCH * HIDDEN;

    // stage h_t [16][512] bf16 (contiguous block) into LDS
    {
      const uint4* src = (const uint4*)(cur + (size_t)b0 * HIDDEN);
      uint4* dst = (uint4*)h_s;
      for (int c = tid; c < 16 * HIDDEN * 2 / 16; c += 128) dst[c] = src[c];
    }
    __syncthreads();

    v8f acc[3];
#pragma unroll
    for (int g = 0; g < 3; ++g)
#pragma unroll
      for (int v = 0; v < 8; ++v) acc[g][v] = 0.0f;

#pragma unroll 4
    for (int kk = 0; kk < HIDDEN / 32; ++kk) {
      const bf16* ap = h_s + (size_t)m * HIDDEN + kk * 32 + hi * 8;
      bf16x8 a0 = *(const bf16x8*)ap;
      bf16x8 a1 = *(const bf16x8*)(ap + 16);
      const v16bf A = mk16(a0, a1);
      const int row = kk * 32 + hi * 8;
#pragma unroll
      for (int g = 0; g < 3; ++g) {
        const bf16* bp = wh_s + ((size_t)(g * HIDDEN + row) * 64 + wave * 16 + m);
        v16bf Bv;
#pragma unroll
        for (int i = 0; i < 8; ++i) { Bv[i] = bp[i * 64]; Bv[i + 8] = bp[(i + 16) * 64]; }
        acc[g] = wmma_bf16(A, Bv, acc[g]);
      }
    }

    // pointwise GRU update; C row M = v + 8*hi -> batch b0+M, col = jcol
#pragma unroll
    for (int v = 0; v < 8; ++v) {
      const int bi = b0 + v + 8 * hi;
      const size_t xrow = ((size_t)t * BATCH + bi) * GATES;
      const float xr = (float)xg[xrow + jcol];
      const float xz = (float)xg[xrow + HIDDEN + jcol];
      const float xn = (float)xg[xrow + 2 * HIDDEN + jcol];
      const float r = sigmoidf_(xr + acc[0][v] + bhr);
      const float z = sigmoidf_(xz + acc[1][v] + bhz);
      const float n = tanhf(xn + r * (acc[2][v] + bhn));
      const float hv = (1.0f - z) * n + z * hold[v];
      hold[v] = hv;
      nxt[(size_t)bi * HIDDEN + jcol] = (bf16)hv;
      if (t == SEQ - 1) hlast[(size_t)bi * HIDDEN + jcol] = hv;
    }

    // one sync per step (h ping-pong): also fences LDS h_s reuse
    group_barrier(cnt, gen, 8);
  }
}

// ---------------- kernel 3: logits = h_last @ Wo + bo ----------------
__global__ void logits_kernel(const float* __restrict__ hlast, const float* __restrict__ Wo,
                              const float* __restrict__ bo, float* __restrict__ out) {
  const int t = threadIdx.x;
  if (t >= BATCH * 2) return;
  const int b = t >> 1, c = t & 1;
  float acc = bo[c];
  const float* hr = hlast + (size_t)b * HIDDEN;
#pragma unroll 8
  for (int k = 0; k < HIDDEN; ++k) acc += hr[k] * Wo[k * 2 + c];
  out[t] = acc;
}

// ---------------- host launcher ----------------
extern "C" void kernel_launch(void* const* d_in, const int* in_sizes, int n_in,
                              void* d_out, int out_size, void* d_ws, size_t ws_size,
                              hipStream_t stream) {
  const int*   ids = (const int*)d_in[0];
  const float* emb = (const float*)d_in[1];
  const float* Wx  = (const float*)d_in[2];
  const float* Wh  = (const float*)d_in[3];
  const float* bx  = (const float*)d_in[4];
  const float* bh  = (const float*)d_in[5];
  const float* Wo  = (const float*)d_in[6];
  const float* bo  = (const float*)d_in[7];
  float* out = (float*)d_out;
  char* ws = (char*)d_ws;

  bf16* Wxb   = (bf16*)(ws + OFF_WXB);
  bf16* Whb   = (bf16*)(ws + OFF_WHB);
  bf16* xg    = (bf16*)(ws + OFF_XG);
  bf16* hbuf  = (bf16*)(ws + OFF_HBUF);
  float* hlast = (float*)(ws + OFF_HLAST);
  int*  sync  = (int*)(ws + OFF_SYNC);

  // h0 = 0 (both ping-pong buffers) and fresh sync state, every call
  (void)hipMemsetAsync(ws + OFF_HBUF, 0, 2 * BATCH * HIDDEN * 2, stream);
  (void)hipMemsetAsync(ws + OFF_SYNC, 0, 1024, stream);

  cvt_weights_kernel<<<512, 256, 0, stream>>>(Wx, Wh, Wxb, Whb);

  xg_gemm_kernel<<<dim3(2048, 6), 128, 0, stream>>>(ids, emb, Wxb, bx, xg);

  const int scan_lds = (3 * HIDDEN * 64 + 16 * HIDDEN) * 2;  // 212992 bytes
  (void)hipFuncSetAttribute((const void*)gru_scan_kernel,
                            hipFuncAttributeMaxDynamicSharedMemorySize, scan_lds);
  gru_scan_kernel<<<32, 128, scan_lds, stream>>>(Whb, xg, bh, hbuf, hlast, sync);

  logits_kernel<<<1, 128, 0, stream>>>(hlast, Wo, bo, out);

  (void)in_sizes; (void)n_in; (void)out_size; (void)ws_size;
}
